// RetainAttention_26396869001739
// MI455X (gfx1250) — compile-verified, matrix operand order checked by reading, not run on hardware
//
#include <hip/hip_runtime.h>

typedef __attribute__((ext_vector_type(16))) _Float16 v16h;
typedef __attribute__((ext_vector_type(8)))  _Float16 v8h;
typedef __attribute__((ext_vector_type(4)))  _Float16 v4h;
typedef __attribute__((ext_vector_type(8)))  float    v8f;

#define B_ 64
#define S_ 1024
#define H_ 256
#define E_ 256

__device__ __forceinline__ float fast_tanh(float x) {
  // tanh(x) = 1 - 2/(exp(2x)+1); exact limits at +-inf via rcp(inf)=0.
  const float ex = __expf(2.0f * x);
  return 1.0f - 2.0f * __builtin_amdgcn_rcpf(ex + 1.0f);
}

// ---------------------------------------------------------------------------
// Kernel A: per-row gate weights + reciprocal suffix-sum denominator.
//   w[b,j]    = exp(alpha[b,j,:].Wa + ba) * mask[b,j]
//   rden[b,i] = 1 / (sum_{j>=i} w[b,j] + 1e-10)
// ---------------------------------------------------------------------------
__global__ __launch_bounds__(256) void ra_alpha_kernel(
    const float* __restrict__ alpha, const float* __restrict__ mask,
    const float* __restrict__ Wa,    const float* __restrict__ ba,
    float* __restrict__ w_out,       float* __restrict__ rden_out)
{
  __shared__ float wlds[S_];
  __shared__ float segsum[256];

  const int b    = blockIdx.x;
  const int tid  = threadIdx.x;
  const int lane = tid & 31;
  const int wave = tid >> 5;          // 0..7

  float wa[8];
  #pragma unroll
  for (int m = 0; m < 8; ++m) wa[m] = Wa[lane + 32 * m];
  const float bav = ba[0];

  const float* arow_base = alpha + (size_t)b * S_ * H_;
  for (int iter = 0; iter < S_ / 8; ++iter) {
    const int j = iter * 8 + wave;
    const float* arow = arow_base + (size_t)j * H_;
    float p = 0.f;
    #pragma unroll
    for (int m = 0; m < 8; ++m) p += arow[lane + 32 * m] * wa[m];
    #pragma unroll
    for (int off = 16; off > 0; off >>= 1) p += __shfl_down(p, off, 32);
    if (lane == 0) {
      const float w = __expf(p + bav) * mask[b * S_ + j];
      wlds[j] = w;
      w_out[b * S_ + j] = w;
    }
  }
  __syncthreads();

  // Suffix scan: 4 contiguous elements/thread + Hillis-Steele over 256 sums.
  float local[4];
  float ssum = 0.f;
  #pragma unroll
  for (int q = 3; q >= 0; --q) { local[q] = wlds[tid * 4 + q]; ssum += local[q]; }
  segsum[tid] = ssum;
  __syncthreads();
  #pragma unroll
  for (int off = 1; off < 256; off <<= 1) {
    const float v = (tid + off < 256) ? segsum[tid + off] : 0.f;
    __syncthreads();
    segsum[tid] += v;
    __syncthreads();
  }
  const float tail = segsum[tid] - ssum;
  float run = tail + 1e-10f;
  #pragma unroll
  for (int q = 3; q >= 0; --q) {
    run += local[q];
    rden_out[b * S_ + tid * 4 + q] = 1.0f / run;   // reciprocal: epilogue multiplies
  }
}

// ---------------------------------------------------------------------------
// Prepass: beta f32 -> f16 (removes cvt work + halves bytes in the hot loop).
// ---------------------------------------------------------------------------
__global__ __launch_bounds__(256) void ra_cvt_kernel(
    const float* __restrict__ src, _Float16* __restrict__ dst)
{
  const size_t i = (size_t)blockIdx.x * blockDim.x + threadIdx.x;  // 4 elems each
  const float4 v = ((const float4*)src)[i];
  v4h h;
  h[0] = (_Float16)v.x; h[1] = (_Float16)v.y;
  h[2] = (_Float16)v.z; h[3] = (_Float16)v.w;
  ((v4h*)dst)[i] = h;
}

// ---------------------------------------------------------------------------
// Kernel B: WMMA beta-gate GEMM fused with reverse (suffix) scan over S.
// Block = (batch b, 64-column e-tile). 4 waves; wave owns 16 columns.
// Full A-operand set double-buffered across tiles: next tile's 16 b128 loads
// are issued right after the WMMAs consume the current set, so their latency
// hides behind the epilogue. Dual accumulators break the WMMA D-dep chain.
// ---------------------------------------------------------------------------
template <bool F16PATH>
__global__ __launch_bounds__(128) void ra_main_kernel(
    const float* __restrict__ beta,  const _Float16* __restrict__ beta16,
    const float* __restrict__ embed,
    const float* __restrict__ Wb,    const float* __restrict__ bb,
    const float* __restrict__ w_in,  const float* __restrict__ rden_in,
    float* __restrict__ out)
{
  const int b    = blockIdx.x;
  const int et   = blockIdx.y;
  const int wave = threadIdx.x >> 5;
  const int lane = threadIdx.x & 31;
  const int n    = lane & 15;               // column / A-row within tile
  const int half = lane >> 4;               // ISA lane-half (K split)
  const int e    = et * 64 + wave * 16 + n;

  // Preload B fragments (Wb slice, f32 -> f16 once; loop-invariant).
  // 16-bit operand layout: k(elem,half) = (elem<8?0:16) + 8*half + elem%8.
  const float* wbrow = Wb + (size_t)e * H_;
  v16h bfrag[8];
  #pragma unroll
  for (int kk = 0; kk < 8; ++kk) {
    const int h0 = kk * 32;
    #pragma unroll
    for (int v = 0; v < 8; ++v) {
      const int k = ((v < 4) ? 0 : 16) + 8 * half + 2 * (v & 3);
      const float2 t = *(const float2*)(wbrow + h0 + k);
      bfrag[kk][2 * v]     = (_Float16)t.x;
      bfrag[kk][2 * v + 1] = (_Float16)t.y;
    }
  }
  const float bbe = bb[e];

  // Walking pointers for the reverse j-tile loop.
  const int j_top = S_ - 16;                 // first tile processed: j0 = S-16
  const float*    arow32 = beta   + ((size_t)b * S_ + j_top + n) * H_;
  const _Float16* aptr   = beta16 + ((size_t)b * S_ + j_top + n) * H_;
  const float*    erow   = embed  + ((size_t)b * S_ + j_top) * E_ + e;
  float*          orow   = out    + ((size_t)b * S_ + j_top) * E_ + e;
  const float*    wrow   = w_in    + b * S_ + j_top;
  const float*    drow   = rden_in + b * S_ + j_top;

  // F16 path: prime the full A-operand set for the first tile.
  v16h af[8];
  if (F16PATH) {
    #pragma unroll
    for (int kk = 0; kk < 8; ++kk) {
      const int h0 = kk * 32;
      const v8h lo = *(const v8h*)(aptr + h0 + 8 * half);        // k = 8h..+7
      const v8h hi = *(const v8h*)(aptr + h0 + 16 + 8 * half);   // k = 16+8h..+7
      #pragma unroll
      for (int q = 0; q < 8; ++q) { af[kk][q] = lo[q]; af[kk][8 + q] = hi[q]; }
    }
  }

  float carry = 0.f;                         // suffix carry from higher j-tiles
  for (int tile = S_ / 16 - 1; tile >= 0; --tile) {
    v8f acc0 = {}, acc1 = {};
    if (F16PATH) {
      #pragma unroll
      for (int kk = 0; kk < 8; ++kk) {
        if (kk & 1)
          acc1 = __builtin_amdgcn_wmma_f32_16x16x32_f16(
              false, af[kk], false, bfrag[kk], (short)0, acc1, false, false);
        else
          acc0 = __builtin_amdgcn_wmma_f32_16x16x32_f16(
              false, af[kk], false, bfrag[kk], (short)0, acc0, false, false);
      }
      // Refill af[] with the NEXT tile's operands; latency hides behind the
      // epilogue below (independent of these loads).
      aptr -= (size_t)16 * H_;
      if (tile > 0) {
        __builtin_prefetch(aptr - (size_t)16 * H_, 0, 0);  // tile after next
        #pragma unroll
        for (int kk = 0; kk < 8; ++kk) {
          const int h0 = kk * 32;
          const v8h lo = *(const v8h*)(aptr + h0 + 8 * half);
          const v8h hi = *(const v8h*)(aptr + h0 + 16 + 8 * half);
          #pragma unroll
          for (int q = 0; q < 8; ++q) { af[kk][q] = lo[q]; af[kk][8 + q] = hi[q]; }
        }
      }
    } else {
      #pragma unroll
      for (int kk = 0; kk < 8; ++kk) {
        const int h0 = kk * 32;
        v16h afrag;
        #pragma unroll
        for (int v = 0; v < 8; ++v) {
          const int k = ((v < 4) ? 0 : 16) + 8 * half + 2 * (v & 3);
          const float2 t = *(const float2*)(arow32 + h0 + k);
          afrag[2 * v]     = (_Float16)t.x;
          afrag[2 * v + 1] = (_Float16)t.y;
        }
        if (kk & 1)
          acc1 = __builtin_amdgcn_wmma_f32_16x16x32_f16(
              false, afrag, false, bfrag[kk], (short)0, acc1, false, false);
        else
          acc0 = __builtin_amdgcn_wmma_f32_16x16x32_f16(
              false, afrag, false, bfrag[kk], (short)0, acc0, false, false);
      }
      arow32 -= (size_t)16 * H_;
      if (tile > 0) __builtin_prefetch(arow32, 0, 0);
    }

    // Epilogue: C layout — VGPR r holds row (r + 8*half), column n.
    float s[8];
    #pragma unroll
    for (int r = 0; r < 8; ++r) {
      const int ro = r + 8 * half;
      const float g = fast_tanh(acc0[r] + acc1[r] + bbe);
      s[r] = g * erow[(size_t)ro * E_] * wrow[ro];
    }
    #pragma unroll
    for (int r = 6; r >= 0; --r) s[r] += s[r + 1];      // in-lane suffix
    const float hi_total = __shfl(s[0], n + 16, 32);    // rows 8-15 total
    const float add = (half == 0) ? hi_total : 0.f;
    const float tile_total = __shfl(s[0] + add, n, 32); // broadcast from lower half

    #pragma unroll
    for (int r = 0; r < 8; ++r) {
      const int ro = r + 8 * half;
      orow[(size_t)ro * E_] = (s[r] + add + carry) * drow[ro];
    }
    carry += tile_total;

    erow -= (size_t)16 * E_;
    orow -= (size_t)16 * E_;
    wrow -= 16;
    drow -= 16;
  }
}

extern "C" void kernel_launch(void* const* d_in, const int* in_sizes, int n_in,
                              void* d_out, int out_size, void* d_ws, size_t ws_size,
                              hipStream_t stream) {
  const float* alpha = (const float*)d_in[0];
  const float* beta  = (const float*)d_in[1];
  const float* embed = (const float*)d_in[2];
  const float* mask  = (const float*)d_in[3];
  const float* Wb    = (const float*)d_in[4];
  const float* bb    = (const float*)d_in[5];
  const float* Wa    = (const float*)d_in[6];
  const float* ba    = (const float*)d_in[7];
  float* outp = (float*)d_out;

  float* w_buf    = (float*)d_ws;                 // [B,S]
  float* rden_buf = w_buf + (size_t)B_ * S_;      // [B,S]
  const size_t head_bytes = (size_t)2 * B_ * S_ * sizeof(float);     // 512 KB
  const size_t b16_bytes  = (size_t)B_ * S_ * H_ * sizeof(_Float16); // 32 MB
  const bool f16ok = ws_size >= head_bytes + b16_bytes;
  _Float16* beta16 = (_Float16*)((char*)d_ws + head_bytes);

  ra_alpha_kernel<<<dim3(B_), dim3(256), 0, stream>>>(
      alpha, mask, Wa, ba, w_buf, rden_buf);

  if (f16ok) {
    const size_t n4 = (size_t)B_ * S_ * H_ / 4;   // 4 elems per thread
    ra_cvt_kernel<<<dim3((unsigned)(n4 / 256)), dim3(256), 0, stream>>>(beta, beta16);
    ra_main_kernel<true><<<dim3(B_, E_ / 64), dim3(128), 0, stream>>>(
        beta, beta16, embed, Wb, bb, w_buf, rden_buf, outp);
  } else {
    ra_main_kernel<false><<<dim3(B_, E_ / 64), dim3(128), 0, stream>>>(
        beta, beta16, embed, Wb, bb, w_buf, rden_buf, outp);
  }
}
